// MPIS_static_33792802685824
// MI455X (gfx1250) — matrix operand order repaired and checked by reading, not compiled, not run on hardware
//
#include <hip/hip_runtime.h>

#define VTH 1.0f
#define KC 64           // K-chunk staged in LDS per double-buffer slot
#define WAVES 4         // waves per workgroup (wave32)
#define NT 4            // 16-wide N tiles per wave

typedef _Float16 h16;
typedef __attribute__((ext_vector_type(16))) _Float16 v16h;
typedef __attribute__((ext_vector_type(8)))  float    v8f;
typedef __attribute__((ext_vector_type(4)))  unsigned int u32x4;
typedef __attribute__((ext_vector_type(8)))  int      i32x8;
typedef __attribute__((ext_vector_type(4)))  int      i32x4;

#if defined(__HIP_DEVICE_COMPILE__) && defined(__gfx1250__) && \
    __has_builtin(__builtin_amdgcn_tensor_load_to_lds) && \
    __has_builtin(__builtin_amdgcn_s_wait_tensorcnt)
#define USE_TDM 1
#else
#define USE_TDM 0
#endif

// ---------------------------------------------------------------------------
// TDM 2D: stage 16 x KC contiguous fp32 weight slab (row pitch rowLen elems)
// into LDS with a 1-dword pad per 64 dwords -> float[16][KC+1] layout.
// OOB in K (validCols) / M (validRows) reads return zero.
// ---------------------------------------------------------------------------
__device__ __forceinline__ void tdm_stage_2d(const float* gsrc, unsigned ldsOff,
                                             int rowLen, int validCols, int validRows)
{
#if USE_TDM
    unsigned long long ga = (unsigned long long)(const void*)gsrc;
    unsigned td0 = (unsigned)(validCols < 0 ? 0 : validCols);
    unsigned td1 = (unsigned)(validRows < 0 ? 0 : validRows);

    u32x4 g0;
    g0[0] = 1u;                                                  // count=1 valid descriptor
    g0[1] = ldsOff;                                              // lds_addr
    g0[2] = (unsigned)(ga & 0xffffffffull);                      // global_addr[31:0]
    g0[3] = (unsigned)((ga >> 32) & 0x1ffffffull) | (2u << 30);  // addr[56:32] | type=2

    i32x8 g1;
    // data_size=4B, pad_enable, pad_interval=5 (64 dwords), pad_amount=0 (1 dword)
    g1[0] = (int)((2u << 16) | (1u << 20) | (5u << 22));
    g1[1] = (int)((td0 & 0xffffu) << 16);                        // tensor_dim0 lo16
    g1[2] = (int)((td0 >> 16) | ((td1 & 0xffffu) << 16));        // dim0 hi | dim1 lo
    g1[3] = (int)((td1 >> 16) | ((unsigned)KC << 16));           // dim1 hi | tile_dim0=KC
    g1[4] = (int)16u;                                            // tile_dim1=16, tile_dim2=0
    g1[5] = (int)(unsigned)rowLen;                               // tensor_dim0_stride lo32
    g1[6] = 0;
    g1[7] = 0;

    i32x4 gz = {0, 0, 0, 0};
#if __clang_major__ >= 23
    i32x8 gz8 = {0, 0, 0, 0, 0, 0, 0, 0};
    __builtin_amdgcn_tensor_load_to_lds(g0, g1, gz, gz, gz8, 0);
#else
    __builtin_amdgcn_tensor_load_to_lds(g0, g1, gz, gz, 0);
#endif
#else
    (void)gsrc; (void)ldsOff; (void)rowLen; (void)validCols; (void)validRows;
#endif
}

// ---------------------------------------------------------------------------
// TDM 3D strided: stage 16 rows x KC elements where consecutive elements are
// elemStride floats apart and rows are rowStride floats apart (rs-major
// weight gather). Same padded [16][KC+1] LDS layout.
//   x: 1 element; y: KC taps @ tensor_dim0_stride; z: 16 rows @ dim1_stride.
// ---------------------------------------------------------------------------
__device__ __forceinline__ void tdm_stage_strided(const float* gsrc, unsigned ldsOff,
                                                  int elemStride, int rowStride)
{
#if USE_TDM
    unsigned long long ga = (unsigned long long)(const void*)gsrc;
    u32x4 g0;
    g0[0] = 1u;
    g0[1] = ldsOff;
    g0[2] = (unsigned)(ga & 0xffffffffull);
    g0[3] = (unsigned)((ga >> 32) & 0x1ffffffull) | (2u << 30);

    i32x8 g1;
    g1[0] = (int)((2u << 16) | (1u << 20) | (5u << 22));     // 4B, pad 1 dword / 64
    g1[1] = (int)(1u << 16);                                 // tensor_dim0 = 1
    g1[2] = (int)((unsigned)KC << 16);                       // tensor_dim1 = KC
    g1[3] = (int)(1u << 16);                                 // tile_dim0 = 1
    g1[4] = (int)((unsigned)KC | (16u << 16));               // tile_dim1=KC, tile_dim2=16
    g1[5] = (int)(unsigned)elemStride;                       // tensor_dim0_stride lo32
    g1[6] = (int)(((unsigned)rowStride & 0xffffu) << 16);    // dim0 hi=0 | dim1_stride lo16
    g1[7] = (int)((unsigned)rowStride >> 16);                // dim1_stride bits 47:16
    i32x4 g2 = {16, 0, 0, 0};                                // tensor_dim2 = 16
    i32x4 gz = {0, 0, 0, 0};
#if __clang_major__ >= 23
    i32x8 gz8 = {0, 0, 0, 0, 0, 0, 0, 0};
    __builtin_amdgcn_tensor_load_to_lds(g0, g1, g2, gz, gz8, 0);
#else
    __builtin_amdgcn_tensor_load_to_lds(g0, g1, g2, gz, 0);
#endif
#else
    (void)gsrc; (void)ldsOff; (void)elemStride; (void)rowStride;
#endif
}

// ---------------------------------------------------------------------------
// Implicit-GEMM convolution via v_wmma_f32_16x16x32_f16.
//   MODE 0: generic conv (any Ci), ci-major K, contiguous [Co][K] weights
//   MODE 1: convT 3x3 s2 p1 op1, rs-major K (requires Ci%KC==0), w [Ci,Co,3,3]
//   MODE 2: pointwise/GEMM, weights [Co,Ci]
//   MODE 3: conv 3x3, rs-major K (requires Ci%KC==0), w [Co,Ci,3,3]
// rs-major K (k = (r*KW+s)*Ci + ci) puts a whole KC chunk inside one (r,s)
// tap: window/parity tests hoist to once per chunk, and the B gather is one
// strided load + cvt per element. Invalid (OOB-N) columns compute garbage at
// safe addresses and are not stored. Epilogue fuses bias + BN + clamp[0,VTH].
// ---------------------------------------------------------------------------
template <int MODE>
__global__ __launch_bounds__(WAVES * 32)
void conv_wmma(const float* __restrict__ in, const float* __restrict__ wgt,
               const float* __restrict__ bias, const float* __restrict__ bng,
               const float* __restrict__ bnb, float* __restrict__ out,
               int B, int Ci, int Hi, int Wi, int Co, int Ho, int Wo,
               int KH, int KW, int stride, int pad, int doClamp)
{
    __shared__ float lw[2][16][KC + 1];

    const int tid = threadIdx.x;
    const int wave = tid >> 5;
    const int lane = tid & 31;
    const int laneCol = lane & 15;
    const int hiHalf = lane >> 4;

    const int khw = KH * KW;
    const int K = Ci * khw;
    const int HiWi = Hi * Wi;
    const int coBase = blockIdx.x * 16;
    const int HoWo = Ho * Wo;
    const int Ntot = B * HoWo;

    // Per-tile invariants.
    const float* inB[NT];
    int ph[NT], pw[NT];
    int hh[NT], ww[NT], nn[NT];
    bool cv[NT];
#pragma unroll
    for (int t = 0; t < NT; ++t) {
        int col = (blockIdx.y * (WAVES * NT) + wave * NT + t) * 16 + laneCol;
        cv[t] = (col < Ntot);
        int c = cv[t] ? col : 0;
        nn[t] = c / HoWo;
        int r = c - nn[t] * HoWo;
        hh[t] = r / Wo;
        ww[t] = r - hh[t] * Wo;
        inB[t] = in + (size_t)nn[t] * Ci * HiWi;
        if (MODE == 0 || MODE == 3) { ph[t] = hh[t] * stride - pad; pw[t] = ww[t] * stride - pad; }
        else if (MODE == 1)         { ph[t] = hh[t] + 1;            pw[t] = ww[t] + 1;            }
        else                        { ph[t] = hh[t] * Wi + ww[t];   pw[t] = 0;                    }
    }

    // MODE 0: persistent per-lane im2col cursor (ci-major K).
    int cc = 0, rr = 0, ss = 0, q32 = 0, dr32 = 0, ds32 = 0;
    if (MODE == 0) {
        int k0 = hiHalf * 16;
        cc = k0 / khw;
        int rs = k0 - cc * khw;
        rr = rs / KW;
        ss = rs - rr * KW;
        q32 = 32 / khw;
        int rem = 32 - q32 * khw;
        dr32 = rem / KW;
        ds32 = rem - dr32 * KW;
    }

    v8f acc[NT];
#pragma unroll
    for (int t = 0; t < NT; ++t) { v8f z = {}; acc[t] = z; }

    const int nChunks = (K + KC - 1) / KC;

    // Staging cursor for rs-major modes (advances once per stage call).
    int stCi = 0, stR = 0, stS = 0;

    auto stage = [&](int chunkIdx, int buf) {
        unsigned ldsOff = (unsigned)(unsigned long long)&lw[buf][0][0];
        if (MODE == 0 || MODE == 2) {
            int kc = chunkIdx * KC;
#if USE_TDM
            if (wave == 0)
                tdm_stage_2d(wgt + (size_t)coBase * K + kc, ldsOff, K, K - kc, Co - coBase);
#else
            for (int idx = tid; idx < 16 * KC; idx += WAVES * 32) {
                int m = idx / KC;
                int j = idx - m * KC;
                int k = kc + j;
                int c = coBase + m;
                float v = 0.f;
                if (k < K && c < Co) v = wgt[(size_t)c * K + k];
                lw[buf][m][j] = v;
            }
#endif
        } else {
            // rs-major: this chunk is tap (stR,stS), channels [stCi, stCi+KC).
            int rsIdx = stR * KW + stS;
            const float* src;
            int eStr, rStr;
            if (MODE == 3) {
                src = wgt + ((size_t)coBase * Ci + stCi) * khw + rsIdx;
                eStr = khw;          // next ci
                rStr = Ci * khw;     // next co
            } else {
                src = wgt + ((size_t)stCi * Co + coBase) * khw + rsIdx;
                eStr = Co * khw;     // next ci
                rStr = khw;          // next co
            }
#if USE_TDM
            if (wave == 0) tdm_stage_strided(src, ldsOff, eStr, rStr);
#else
            for (int idx = tid; idx < 16 * KC; idx += WAVES * 32) {
                int m = idx / KC;
                int j = idx - m * KC;
                lw[buf][m][j] = src[(size_t)j * eStr + (size_t)m * rStr];
            }
#endif
            stCi += KC;
            if (stCi >= Ci) { stCi = 0; if (++stS == KW) { stS = 0; if (++stR == KH) stR = 0; } }
        }
        (void)chunkIdx;
    };

    auto finish = [&]() {
#if USE_TDM
        if (wave == 0) __builtin_amdgcn_s_wait_tensorcnt(0);
#endif
        __syncthreads();
    };

    stage(0, 0);
    finish();

    // Consumer-side chunk state for rs-major modes.
    const float* pB[NT];
    bool okT[NT];
    int ciBase = 0, cR = 0, cS = 0;

    const int nSteps = nChunks * 2;
    for (int m = 0; m < nSteps; ++m) {
        const int chunk = m >> 1;
        const int buf = chunk & 1;

        if ((m & 1) == 0) {
            if (chunk + 1 < nChunks) stage(chunk + 1, buf ^ 1);
            if (MODE == 1 || MODE == 3) {
                if (m > 0) {
                    ciBase += KC;
                    if (ciBase >= Ci) { ciBase = 0; if (++cS == KW) { cS = 0; if (++cR == KH) cR = 0; } }
                }
#pragma unroll
                for (int t = 0; t < NT; ++t) {
                    int hi2, wi2;
                    bool ok;
                    if (MODE == 3) {
                        hi2 = ph[t] + cR;
                        wi2 = pw[t] + cS;
                        ok = (unsigned)hi2 < (unsigned)Hi && (unsigned)wi2 < (unsigned)Wi;
                    } else {
                        int hs = ph[t] - cR, wsd = pw[t] - cS;
                        hi2 = hs >> 1;
                        wi2 = wsd >> 1;
                        ok = !((hs | wsd) & 1) &&
                             (unsigned)hi2 < (unsigned)Hi && (unsigned)wi2 < (unsigned)Wi;
                    }
                    okT[t] = ok;
                    pB[t] = inB[t] + (size_t)ciBase * HiWi + (ok ? (hi2 * Wi + wi2) : 0);
                }
            }
        }

        const int kl = (m & 1) * 32;

        // A-fragment (16x32 f16): lanes 0-15 hold M=lane, K {0..15};
        // lanes 16-31 hold M=lane-16, K {16..31}.
        v16h Af;
#pragma unroll
        for (int e = 0; e < 16; ++e)
            Af[e] = (h16)lw[buf][laneCol][kl + hiHalf * 16 + e];

        const int klocal = kl + hiHalf * 16;   // this lane's first B-row in chunk
        const int kg = m * 32 + hiHalf * 16;   // absolute (modes 0/2)

#pragma unroll
        for (int t = 0; t < NT; ++t) {
            v16h Bf;
            if (MODE == 1 || MODE == 3) {
                const float* p = pB[t] + (size_t)klocal * HiWi;
#pragma unroll
                for (int j = 0; j < 16; ++j) {
                    float v = p[(size_t)j * HiWi];
                    Bf[j] = (h16)(okT[t] ? v : 0.f);
                }
            } else if (MODE == 2) {
                const float* p = inB[t] + ph[t];
#pragma unroll
                for (int j = 0; j < 16; ++j) {
                    int k = kg + j;
                    int kk = (k < K) ? k : 0;
                    float v = p[(size_t)kk * HiWi];
                    Bf[j] = (h16)((k < K) ? v : 0.f);
                }
            } else {
                int c2 = cc, r2 = rr, s2 = ss;
#pragma unroll
                for (int j = 0; j < 16; ++j) {
                    float v = 0.f;
                    int hi2 = ph[t] + r2;
                    int wi2 = pw[t] + s2;
                    if ((unsigned)hi2 < (unsigned)Hi &&
                        (unsigned)wi2 < (unsigned)Wi && c2 < Ci)
                        v = inB[t][c2 * HiWi + hi2 * Wi + wi2];
                    Bf[j] = (h16)v;
                    if (++s2 == KW) { s2 = 0; if (++r2 == KH) { r2 = 0; ++c2; } }
                }
            }
            acc[t] = __builtin_amdgcn_wmma_f32_16x16x32_f16(
                false, Af, false, Bf, (short)0, acc[t], false, false);
        }

        if (MODE == 0) {   // advance cursor by 32 K-rows (no divides)
            cc += q32;
            ss += ds32; if (ss >= KW) { ss -= KW; ++rr; }
            rr += dr32; if (rr >= KH) { rr -= KH; ++cc; }
        }

        if ((m & 1) == 1 && chunk + 1 < nChunks)
            finish();
    }

    // Epilogue: C/D VGPR i holds M=i (lanes 0-15) / M=i+8 (lanes 16-31).
#pragma unroll
    for (int t = 0; t < NT; ++t) {
        if (!cv[t]) continue;
#pragma unroll
        for (int i = 0; i < 8; ++i) {
            int cch = coBase + hiHalf * 8 + i;
            if (cch >= Co) continue;
            float v = acc[t][i];
            if (bias) v += bias[cch];
            if (bng)  v = v * bng[cch] + bnb[cch];
            if (doClamp) v = fminf(fmaxf(v, 0.f), VTH);
            out[(((size_t)nn[t] * Co + cch) * Ho + hh[t]) * Wo + ww[t]] = v;
        }
    }
}

// ---------------------------- elementwise helpers ---------------------------
__global__ void k_zero(float* p, int n)
{
    int i = blockIdx.x * 256 + threadIdx.x;
    if (i < n) p[i] = 0.f;
}

__global__ void k_addclamp(const float* __restrict__ a, const float* __restrict__ b,
                           float* __restrict__ o, int n)
{
    int i = blockIdx.x * 256 + threadIdx.x;
    if (i < n) o[i] = fminf(fmaxf(a[i] + b[i], 0.f), VTH);
}

__global__ void k_add(const float* __restrict__ a, const float* __restrict__ b,
                      float* __restrict__ o, int n)
{
    int i = blockIdx.x * 256 + threadIdx.x;
    if (i < n) o[i] = a[i] + b[i];
}

__global__ void k_pool2x2(const float* __restrict__ in, float* __restrict__ out,
                          int n, int C, int Ho, int Wo)
{
    int i = blockIdx.x * 256 + threadIdx.x;
    if (i >= n) return;
    int wo = i % Wo;
    int t = i / Wo;
    int ho = t % Ho; t /= Ho;
    int c = t % C;
    int nb = t / C;
    int Wi = Wo * 2, Hi = Ho * 2;
    const float* p = in + (((size_t)nb * C + c) * Hi + ho * 2) * Wi + wo * 2;
    out[i] = 0.25f * (p[0] + p[1] + p[Wi] + p[Wi + 1]);
}

// ------------------------------- orchestration ------------------------------
extern "C" void kernel_launch(void* const* d_in, const int* in_sizes, int n_in,
                              void* d_out, int out_size, void* d_ws, size_t ws_size,
                              hipStream_t stream)
{
    (void)in_sizes; (void)n_in; (void)out_size; (void)ws_size;

    // d_in[0] = x; params flattened as a jax pytree (sorted keys, bn {b,g}).
    const float* x        = (const float*)d_in[0];
    const float* cls_b    = (const float*)d_in[1];
    const float* cls_w    = (const float*)d_in[2];
    const float* di_b1    = (const float*)d_in[3];
    const float* di_b2    = (const float*)d_in[4];
    const float* di_bn1_b = (const float*)d_in[5];
    const float* di_bn1_g = (const float*)d_in[6];
    const float* di_bn2_b = (const float*)d_in[7];
    const float* di_bn2_g = (const float*)d_in[8];
    const float* di_w1    = (const float*)d_in[9];
    const float* di_w2    = (const float*)d_in[10];
    const float* ds_b     = (const float*)d_in[11];
    const float* ds_bn_b  = (const float*)d_in[12];
    const float* ds_bn_g  = (const float*)d_in[13];
    const float* ds_w     = (const float*)d_in[14];
    const float* inc_bn_b = (const float*)d_in[15];
    const float* inc_bn_g = (const float*)d_in[16];
    const float* inc_w    = (const float*)d_in[17];
    const float* s1_b     = (const float*)d_in[18];
    const float* s1_bn_b  = (const float*)d_in[19];
    const float* s1_bn_g  = (const float*)d_in[20];
    const float* s1_w     = (const float*)d_in[21];
    const float* s1n_b    = (const float*)d_in[22];
    const float* s1n_bn_b = (const float*)d_in[23];
    const float* s1n_bn_g = (const float*)d_in[24];
    const float* s1n_w    = (const float*)d_in[25];
    const float* s2_b     = (const float*)d_in[26];
    const float* s2_bn_b  = (const float*)d_in[27];
    const float* s2_bn_g  = (const float*)d_in[28];
    const float* s2_w     = (const float*)d_in[29];
    const float* s2n_b    = (const float*)d_in[30];
    const float* s2n_bn_b = (const float*)d_in[31];
    const float* s2n_bn_g = (const float*)d_in[32];
    const float* s2n_w    = (const float*)d_in[33];
    const float* tr_w     = (const float*)d_in[34];
    const float* trn_w    = (const float*)d_in[35];
    const float* wx_b     = (const float*)d_in[36];
    const float* wx_bn_b  = (const float*)d_in[37];
    const float* wx_bn_g  = (const float*)d_in[38];
    const float* wx_w     = (const float*)d_in[39];
    const float* wxn_b    = (const float*)d_in[40];
    const float* wxn_bn_b = (const float*)d_in[41];
    const float* wxn_bn_g = (const float*)d_in[42];
    const float* wxn_w    = (const float*)d_in[43];

    const int B = 64;
    float* ws = (float*)d_ws;
    size_t off = 0;
    auto alloc = [&](size_t nElems) { float* p = ws + off; off += nElems; return p; };

    float* h1   = alloc((size_t)B * 64 * 64 * 64);   // [B,64,64,64]
    float* h    = alloc((size_t)B * 64 * 32 * 32);   // [B,64,32,32]
    float* inj0 = alloc((size_t)B * 64 * 32 * 32);
    float* hn   = alloc((size_t)B * 64 * 16 * 16);
    float* inj1 = alloc((size_t)B * 64 * 16 * 16);
    float* a0   = alloc((size_t)B * 64 * 32 * 32);
    float* a1   = alloc((size_t)B * 64 * 16 * 16);
    float* ts1  = alloc((size_t)B * 128 * 32 * 32);  // branch-s stage1
    float* ts2  = alloc((size_t)B * 256 * 16 * 16);  // branch-s stage2 / z0
    float* t0   = alloc((size_t)B * 64 * 32 * 32);
    float* tn1  = alloc((size_t)B * 128 * 16 * 16);
    float* tn2  = alloc((size_t)B * 256 * 8 * 8);    // branch-n stage2 / z1
    float* t1   = alloc((size_t)B * 64 * 16 * 16);
    float* zds  = alloc((size_t)B * 256 * 8 * 8);
    float* zsum = alloc((size_t)B * 256 * 8 * 8);
    float* zfin = alloc((size_t)B * 256 * 8 * 8);

    auto convL = [&](int mode, const float* in_, const float* w_, const float* bias_,
                     const float* g_, const float* b_, float* out_,
                     int Ci, int Hi, int Wi, int Co, int Ho, int Wo,
                     int KH, int KW, int st, int pd, int cl) {
        int N = B * Ho * Wo;
        dim3 grid((Co + 15) / 16, (N + WAVES * NT * 16 - 1) / (WAVES * NT * 16));
        dim3 block(WAVES * 32);
        if (mode == 0)
            conv_wmma<0><<<grid, block, 0, stream>>>(in_, w_, bias_, g_, b_, out_,
                B, Ci, Hi, Wi, Co, Ho, Wo, KH, KW, st, pd, cl);
        else if (mode == 1)
            conv_wmma<1><<<grid, block, 0, stream>>>(in_, w_, bias_, g_, b_, out_,
                B, Ci, Hi, Wi, Co, Ho, Wo, KH, KW, st, pd, cl);
        else if (mode == 2)
            conv_wmma<2><<<grid, block, 0, stream>>>(in_, w_, bias_, g_, b_, out_,
                B, Ci, Hi, Wi, Co, Ho, Wo, KH, KW, st, pd, cl);
        else
            conv_wmma<3><<<grid, block, 0, stream>>>(in_, w_, bias_, g_, b_, out_,
                B, Ci, Hi, Wi, Co, Ho, Wo, KH, KW, st, pd, cl);
    };
    auto g1d = [](int n) { return dim3((n + 255) / 256); };

    // ---- downsample_init ----
    convL(0, x,  di_w1, di_b1, di_bn1_g, di_bn1_b, h1, 3,  64, 64, 64, 64, 64, 3, 3, 1, 1, 1);
    convL(3, h1, di_w2, di_b2, di_bn2_g, di_bn2_b, h,  64, 64, 64, 64, 32, 32, 3, 3, 2, 1, 1);

    // ---- constant injections ----
    convL(3, h, wx_w, wx_b, wx_bn_g, wx_bn_b, inj0, 64, 32, 32, 64, 32, 32, 3, 3, 1, 1, 1);
    {
        int n = B * 64 * 16 * 16;
        k_pool2x2<<<g1d(n), 256, 0, stream>>>(h, hn, n, 64, 16, 16);
    }
    convL(3, hn, wxn_w, wxn_b, wxn_bn_g, wxn_bn_b, inj1, 64, 16, 16, 64, 16, 16, 3, 3, 1, 1, 1);

    // ---- equilibrium iteration ----
    const int n0 = B * 64 * 32 * 32;
    const int n1 = B * 64 * 16 * 16;
    k_zero<<<g1d(n0), 256, 0, stream>>>(a0, n0);
    k_zero<<<g1d(n1), 256, 0, stream>>>(a1, n1);

    for (int step = 0; step < 8; ++step) {
        // branch s: s1 -> s2(stride2) -> convT(tr_w)
        convL(3, a0,  s1_w, s1_b, s1_bn_g, s1_bn_b, ts1, 64,  32, 32, 128, 32, 32, 3, 3, 1, 1, 1);
        convL(3, ts1, s2_w, s2_b, s2_bn_g, s2_bn_b, ts2, 128, 32, 32, 256, 16, 16, 3, 3, 2, 1, 1);
        convL(1, ts2, tr_w, nullptr, nullptr, nullptr, t0, 256, 16, 16, 64, 32, 32, 3, 3, 2, 1, 0);
        // branch n
        convL(3, a1,  s1n_w, s1n_b, s1n_bn_g, s1n_bn_b, tn1, 64,  16, 16, 128, 16, 16, 3, 3, 1, 1, 1);
        convL(3, tn1, s2n_w, s2n_b, s2n_bn_g, s2n_bn_b, tn2, 128, 16, 16, 256, 8,  8,  3, 3, 2, 1, 1);
        convL(1, tn2, trn_w, nullptr, nullptr, nullptr, t1, 256, 8, 8, 64, 16, 16, 3, 3, 2, 1, 0);
        // a = dbrelu(t + inj)
        k_addclamp<<<g1d(n0), 256, 0, stream>>>(t0, inj0, a0, n0);
        k_addclamp<<<g1d(n1), 256, 0, stream>>>(t1, inj1, a1, n1);
    }

    // ---- equilibrium outputs ----
    convL(3, a0,  s1_w,  s1_b,  s1_bn_g,  s1_bn_b,  ts1, 64,  32, 32, 128, 32, 32, 3, 3, 1, 1, 1);
    convL(3, ts1, s2_w,  s2_b,  s2_bn_g,  s2_bn_b,  ts2, 128, 32, 32, 256, 16, 16, 3, 3, 2, 1, 1); // z0
    convL(3, a1,  s1n_w, s1n_b, s1n_bn_g, s1n_bn_b, tn1, 64,  16, 16, 128, 16, 16, 3, 3, 1, 1, 1);
    convL(3, tn1, s2n_w, s2n_b, s2n_bn_g, s2n_bn_b, tn2, 128, 16, 16, 256, 8,  8,  3, 3, 2, 1, 1); // z1

    // ---- fusion head ----
    convL(3, ts2, ds_w, ds_b, ds_bn_g, ds_bn_b, zds, 256, 16, 16, 256, 8, 8, 3, 3, 2, 1, 1);
    {
        int n = B * 256 * 8 * 8;
        k_add<<<g1d(n), 256, 0, stream>>>(zds, tn2, zsum, n);
    }
    convL(2, zsum, inc_w, nullptr, inc_bn_g, inc_bn_b, zfin, 256, 8, 8, 256, 8, 8, 1, 1, 1, 0, 1);

    // ---- classifier: GEMM as a 1x1 conv over Ci = 256*8*8, H=W=1 ----
    convL(2, zfin, cls_w, cls_b, nullptr, nullptr, (float*)d_out,
          256 * 8 * 8, 1, 1, 100, 1, 1, 1, 1, 1, 0, 0);
}
